// SNN_MLP_23407571764104
// MI455X (gfx1250) — 8-bit matrix operand order repaired and checked by reading, compiled, not run on hardware
//
#include <hip/hip_runtime.h>


typedef __attribute__((ext_vector_type(16))) int   v16i;
typedef __attribute__((ext_vector_type(8)))  float v8f;

#define LDS_ROW_STRIDE 144

__device__ __forceinline__ unsigned char f32_to_e4m3(float x) {
    unsigned int u    = __float_as_uint(x);
    unsigned int sign = (u >> 24) & 0x80u;
    unsigned int abs  = u & 0x7fffffffu;
    float a = __uint_as_float(abs);
    if (a >= 448.0f) return (unsigned char)(sign | 0x7Eu);
    if (a < 0.015625f) {
        float m = rintf(a * 512.0f);
        return (unsigned char)(sign | (unsigned int)m);
    }
    int e = (int)(abs >> 23) - 127;
    unsigned int m24  = (abs & 0x7fffffu) | 0x800000u;
    unsigned int mant = m24 >> 20;
    unsigned int rem  = m24 & 0xFFFFFu;
    const unsigned int halfp = 0x80000u;
    mant += (rem > halfp || (rem == halfp && (mant & 1u))) ? 1u : 0u;
    if (mant == 16u) { mant = 8u; ++e; }
    if (e > 8) return (unsigned char)(sign | 0x7Eu);
    return (unsigned char)(sign | ((unsigned int)(e + 7) << 3) | (mant & 7u));
}

__device__ __forceinline__ float e4m3_to_f32(unsigned char c) {
    unsigned int sign = (c & 0x80u) ? 0x80000000u : 0u;
    unsigned int exp  = ((unsigned int)c >> 3) & 0xFu;
    unsigned int mant = (unsigned int)c & 7u;
    float v;
    if (exp == 0u) v = (float)mant * 0.001953125f;
    else           v = __uint_as_float(((exp + 120u) << 23) | (mant << 20));
    return __uint_as_float(__float_as_uint(v) | sign);
}

__global__ __launch_bounds__(256) void quant_fp8_kernel(
    const float4* __restrict__ in, uchar4* __restrict__ out, int n4) {
    int i = blockIdx.x * blockDim.x + threadIdx.x;
    if (i >= n4) return;
    float4 v = in[i];
    uchar4 r;
    r.x = f32_to_e4m3(v.x);
    r.y = f32_to_e4m3(v.y);
    r.z = f32_to_e4m3(v.z);
    r.w = f32_to_e4m3(v.w);
    out[i] = r;
}

__device__ __forceinline__ void stage_B_async(
    const unsigned char* __restrict__ W, int n0, int K, int k0,
    unsigned char* smem_buf, int tid) {
#pragma unroll
    for (int p = 0; p < 2; ++p) {
        const int c   = tid + p * 256;
        const int row = c >> 3;
        const int off = (c & 7) * 16;
        unsigned long long ga =
            (unsigned long long)(size_t)(W + (size_t)(n0 + row) * K + k0 + off);
        unsigned int lds =
            (unsigned int)(size_t)(smem_buf + row * LDS_ROW_STRIDE + off);
        asm volatile("global_load_async_to_lds_b128 %0, %1, off"
                     :: "v"(lds), "v"(ga) : "memory");
    }
}

__device__ __forceinline__ void wait_async_done() {
    asm volatile("s_wait_asynccnt 0x0" ::: "memory");
}

template <bool RELU_FP8>
__global__ __launch_bounds__(256) void gemm_fp8_kernel(
    const unsigned char* __restrict__ A,
    const unsigned char* __restrict__ W,
    unsigned char* __restrict__ out8,
    float*         __restrict__ outf,
    int M, int N, int K) {
    __shared__ __align__(16) unsigned char sB[2][64 * LDS_ROW_STRIDE];

    const int tid  = threadIdx.x;
    const int lane = tid & 31;
    const int wave = tid >> 5;
    const int l    = lane & 15;
    const int half = lane >> 4;
    const int m0   = blockIdx.y * 128 + wave * 16;
    const int n0   = blockIdx.x * 64;

    v8f acc[4];
#pragma unroll
    for (int j = 0; j < 4; ++j) acc[j] = (v8f){};

    const unsigned char* arow = A + (size_t)(m0 + l) * K + half * 8;

    stage_B_async(W, n0, K, 0, &sB[0][0], tid);
    wait_async_done();
    __syncthreads();

    const int KT = K >> 7;
    for (int kt = 0; kt < KT; ++kt) {
        const int k0  = kt << 7;
        const int buf = kt & 1;

        if (kt + 1 < KT)
            stage_B_async(W, n0, K, k0 + 128, &sB[buf ^ 1][0], tid);

        v16i a;
#pragma unroll
        for (int c = 0; c < 8; ++c) {
            uint2 q = *(const uint2*)(arow + k0 + c * 16);
            a[2 * c + 0] = (int)q.x;
            a[2 * c + 1] = (int)q.y;
        }

        v16i b[4];
#pragma unroll
        for (int j = 0; j < 4; ++j) {
            const unsigned char* bp =
                &sB[buf][(j * 16 + l) * LDS_ROW_STRIDE + half * 8];
#pragma unroll
            for (int c = 0; c < 8; ++c) {
                uint2 q = *(const uint2*)(bp + c * 16);
                b[j][2 * c + 0] = (int)q.x;
                b[j][2 * c + 1] = (int)q.y;
            }
        }

#pragma unroll
        for (int j = 0; j < 4; ++j)
            acc[j] = __builtin_amdgcn_wmma_f32_16x16x128_fp8_fp8(
                a, b[j], (short)0, acc[j], false, false);

        wait_async_done();
        __syncthreads();
    }

    const int row0 = m0 + half * 8;
#pragma unroll
    for (int j = 0; j < 4; ++j) {
        const int col = n0 + j * 16 + l;
#pragma unroll
        for (int r = 0; r < 8; ++r) {
            float v = acc[j][r];
            size_t idx = (size_t)(row0 + r) * N + col;
            if (RELU_FP8) {
                unsigned char q = f32_to_e4m3(v);
                if (q & 0x80u) q = 0;
                out8[idx] = q;
            } else {
                outf[idx] = e4m3_to_f32(f32_to_e4m3(v));
            }
        }
    }
}

extern "C" void kernel_launch(void* const* d_in, const int* in_sizes, int n_in,
                              void* d_out, int out_size, void* d_ws, size_t ws_size,
                              hipStream_t stream) {
    (void)in_sizes; (void)n_in; (void)out_size; (void)ws_size;
    const int Bn = 8192, DIN = 1024, DH = 4096, DOUT = 1024;

    const float* x  = (const float*)d_in[0];
    const float* w1 = (const float*)d_in[1];
    const float* w2 = (const float*)d_in[2];
    float* out = (float*)d_out;

    unsigned char* ws  = (unsigned char*)d_ws;
    unsigned char* x8  = ws;
    unsigned char* w1q = x8  + (size_t)Bn   * DIN;
    unsigned char* w2q = w1q + (size_t)DH   * DIN;
    unsigned char* h8  = w2q + (size_t)DOUT * DH;

    const int nx4 = Bn * DIN / 4;
    quant_fp8_kernel<<<nx4 / 256, 256, 0, stream>>>((const float4*)x,  (uchar4*)x8,  nx4);
    const int n14 = DH * DIN / 4;
    quant_fp8_kernel<<<n14 / 256, 256, 0, stream>>>((const float4*)w1, (uchar4*)w1q, n14);
    const int n24 = DOUT * DH / 4;
    quant_fp8_kernel<<<n24 / 256, 256, 0, stream>>>((const float4*)w2, (uchar4*)w2q, n24);

    dim3 g1(DH / 64, Bn / 128);
    gemm_fp8_kernel<true><<<g1, 256, 0, stream>>>(x8, w1q, h8, nullptr, Bn, DH, DIN);

    dim3 g2(DOUT / 64, Bn / 128);
    gemm_fp8_kernel<false><<<g2, 256, 0, stream>>>(h8, w2q, nullptr, out, Bn, DOUT, DH);
}
